// ResidualVectorQuantizer_63445256896889
// MI455X (gfx1250) — matrix operand order repaired and checked by reading, not compiled, not run on hardware
//
#include <hip/hip_runtime.h>
#include <hip/hip_bf16.h>

typedef __attribute__((ext_vector_type(16))) __bf16 v16bf;
typedef __attribute__((ext_vector_type(8)))  float  v8f;

#define RVQ_LEVELS 8
#define RVQ_K      1024
#define RVQ_D      256
#define RVQ_N      65536

#define WAVES_PER_BLOCK 8
#define TOK_PER_BLOCK   (WAVES_PER_BLOCK * 16)
#define KTILES          (RVQ_K / 16)
#define LDS_ROW         272   // 256 bf16 data + 16 bf16 pad (544 B row stride)

// ---------------------------------------------------------------------------
// CDNA5 async global->LDS copy (ASYNCcnt-tracked, no VGPR staging).
// vdst = 32-bit LDS byte address, vaddr = 64-bit global address (GV mode).
// ---------------------------------------------------------------------------
__device__ __forceinline__ void async_copy_b128(unsigned lds_byte_addr,
                                                const void* gptr) {
    asm volatile("global_load_async_to_lds_b128 %0, %1, off"
                 :: "v"(lds_byte_addr),
                    "v"((unsigned long long)(uintptr_t)gptr)
                 : "memory");
}

__device__ __forceinline__ void wait_asynccnt0() {
    asm volatile("s_wait_asynccnt 0x0" ::: "memory");
}

// ---------------------------------------------------------------------------
// Prep: convert fp32 codebooks -> bf16 in workspace, compute c_sq per code.
// One wave per codebook row; each lane handles 8 consecutive dims.
// ---------------------------------------------------------------------------
__global__ __launch_bounds__(128) void rvq_prep_kernel(
    const float* __restrict__ cb, __bf16* __restrict__ cbbf,
    float* __restrict__ csq)
{
    const int wave = threadIdx.x >> 5;
    const int lane = threadIdx.x & 31;
    const int row  = blockIdx.x * 4 + wave;          // 0 .. L*K-1
    const float* src = cb + (size_t)row * RVQ_D + lane * 8;
    __bf16* dst = cbbf + (size_t)row * RVQ_D + lane * 8;

    float4 f0 = *(const float4*)(src);
    float4 f1 = *(const float4*)(src + 4);
    float v[8] = {f0.x, f0.y, f0.z, f0.w, f1.x, f1.y, f1.z, f1.w};

    float ssum = 0.0f;
#pragma unroll
    for (int i = 0; i < 8; ++i) {
        ssum += v[i] * v[i];
        dst[i] = (__bf16)v[i];
    }
#pragma unroll
    for (int m = 1; m < 32; m <<= 1)
        ssum += __shfl_xor(ssum, m, 32);
    if (lane == 0)
        csq[row] = ssum;
}

// ---------------------------------------------------------------------------
// Main RVQ kernel. Block = 8 waves; each wave owns a 16-token M-tile across
// all 8 levels. Per level: residual->bf16 A frags (once), stream the full
// 1024-code bf16 codebook through double-buffered LDS B tiles (async copies),
// accumulate cross products with v_wmma_f32_16x16x32_bf16, lane-local argmin,
// butterfly reduce, gather fp32 codebook row, update fp32 residual registers.
// ---------------------------------------------------------------------------
__global__ __launch_bounds__(256) void rvq_main_kernel(
    const float* __restrict__ x,
    const float* __restrict__ cb,        // fp32 codebooks [L][K][D]
    const __bf16* __restrict__ cbbf,     // bf16 codebooks [L][K][D]
    const float* __restrict__ csq,       // [L][K]
    float* __restrict__ qout,            // [N][D]
    int* __restrict__ codes)             // [N][L]
{
    __shared__ alignas(32) __bf16 bstage[2][16 * LDS_ROW];
    __shared__ int idxLds[WAVES_PER_BLOCK][16];

    const int tid  = threadIdx.x;
    const int wave = tid >> 5;
    const int lane = tid & 31;
    const int half = lane >> 4;          // 0: lanes 0-15, 1: lanes 16-31
    const int l16  = lane & 15;
    const int tokBase = blockIdx.x * TOK_PER_BLOCK + wave * 16;
    const int row  = tokBase + l16;      // token (A-matrix row) for this lane
    const int dbase = half * 8;          // K-offset within each 32-dim chunk

    // Residual in fp32 registers, laid out exactly as the A fragment:
    // r[c][i] = residual[row][32*c + dbase + i + (i>=8 ? 8 : 0)]
    float r[8][16];
#pragma unroll
    for (int c = 0; c < 8; ++c) {
        const float* px = x + (size_t)row * RVQ_D + 32 * c + dbase;
        float4 a0 = *(const float4*)(px);
        float4 a1 = *(const float4*)(px + 4);
        float4 a2 = *(const float4*)(px + 16);
        float4 a3 = *(const float4*)(px + 20);
        r[c][0]=a0.x; r[c][1]=a0.y; r[c][2]=a0.z; r[c][3]=a0.w;
        r[c][4]=a1.x; r[c][5]=a1.y; r[c][6]=a1.z; r[c][7]=a1.w;
        r[c][8]=a2.x; r[c][9]=a2.y; r[c][10]=a2.z; r[c][11]=a2.w;
        r[c][12]=a3.x; r[c][13]=a3.y; r[c][14]=a3.z; r[c][15]=a3.w;
    }

    // Staging geometry: 16 codes x 256 dims bf16 (8 KB), 32 B per thread.
    const int srow = tid >> 4;           // 0..15 (code row within tile)
    const int soff = (tid & 15) * 16;    // bf16 element offset within row

#pragma unroll 1
    for (int lvl = 0; lvl < RVQ_LEVELS; ++lvl) {
        const __bf16* cbl  = cbbf + (size_t)lvl * RVQ_K * RVQ_D;
        const float*  csql = csq + lvl * RVQ_K;

        // Convert residual to bf16 A fragments once per level.
        v16bf abf[8];
#pragma unroll
        for (int c = 0; c < 8; ++c)
#pragma unroll
            for (int i = 0; i < 16; ++i)
                abf[c][i] = (__bf16)r[c][i];

        float bestD[8];
        int   bestI[8];
#pragma unroll
        for (int g = 0; g < 8; ++g) { bestD[g] = 3.4e38f; bestI[g] = 0x7fffffff; }

        // Async-stage ktile 0 into buffer 0.
        {
            const __bf16* s = cbl + srow * RVQ_D + soff;
            unsigned d = (unsigned)(uintptr_t)&bstage[0][srow * LDS_ROW + soff];
            async_copy_b128(d, s);
            async_copy_b128(d + 16, (const char*)s + 16);
        }
        wait_asynccnt0();
        __syncthreads();

#pragma unroll 1
        for (int t = 0; t < KTILES; ++t) {
            // Async-prefetch next tile into the other buffer.
            if (t + 1 < KTILES) {
                const __bf16* s = cbl + (size_t)(t + 1) * 16 * RVQ_D
                                  + srow * RVQ_D + soff;
                unsigned d = (unsigned)(uintptr_t)
                             &bstage[(t + 1) & 1][srow * LDS_ROW + soff];
                async_copy_b128(d, s);
                async_copy_b128(d + 16, (const char*)s + 16);
            }

            const __bf16* bs = &bstage[t & 1][0];
            const float cs = csql[t * 16 + l16];

            v8f acc = {};
#pragma unroll
            for (int grp = 0; grp < 2; ++grp) {
                v16bf b[4];
#pragma unroll
                for (int c = 0; c < 4; ++c)
#pragma unroll
                    for (int s = 0; s < 16; ++s)
                        b[c][s] = bs[l16 * LDS_ROW + (grp * 4 + c) * 32
                                     + half * 8 + s + (s >= 8 ? 8 : 0)];
#pragma unroll
                for (int c = 0; c < 4; ++c)
                    acc = __builtin_amdgcn_wmma_f32_16x16x32_bf16(
                        false, abf[grp * 4 + c], false, b[c],
                        (short)0, acc, false, false);
            }

            const int code = t * 16 + l16;
#pragma unroll
            for (int g = 0; g < 8; ++g) {
                float d = cs - 2.0f * acc[g];
                if (d < bestD[g] || (d == bestD[g] && code < bestI[g])) {
                    bestD[g] = d;
                    bestI[g] = code;
                }
            }

            // Own async copies must have landed in LDS before releasing peers.
            wait_asynccnt0();
            __syncthreads();
        }

        // Butterfly argmin across the 16 lanes of each half (columns).
#pragma unroll
        for (int g = 0; g < 8; ++g) {
            float d = bestD[g];
            int   i = bestI[g];
#pragma unroll
            for (int m = 1; m < 16; m <<= 1) {
                float od = __shfl_xor(d, m, 32);
                int   oi = __shfl_xor(i, m, 32);
                if (od < d || (od == d && oi < i)) { d = od; i = oi; }
            }
            if (l16 == 0)
                idxLds[wave][half * 8 + g] = i;   // token (g + 8*half)
        }
        __syncthreads();

        const int myIdx = idxLds[wave][l16];       // code for token l16
        if (half == 0)
            codes[(size_t)(tokBase + l16) * RVQ_LEVELS + lvl] = myIdx;

        // Gather fp32 codebook row and update residual registers.
        const float* crow = cb + ((size_t)lvl * RVQ_K + myIdx) * RVQ_D;
#pragma unroll
        for (int c = 0; c < 8; ++c) {
            const float* p = crow + 32 * c + dbase;
            float4 a0 = *(const float4*)(p);
            float4 a1 = *(const float4*)(p + 4);
            float4 a2 = *(const float4*)(p + 16);
            float4 a3 = *(const float4*)(p + 20);
            r[c][0]-=a0.x; r[c][1]-=a0.y; r[c][2]-=a0.z; r[c][3]-=a0.w;
            r[c][4]-=a1.x; r[c][5]-=a1.y; r[c][6]-=a1.z; r[c][7]-=a1.w;
            r[c][8]-=a2.x; r[c][9]-=a2.y; r[c][10]-=a2.z; r[c][11]-=a2.w;
            r[c][12]-=a3.x; r[c][13]-=a3.y; r[c][14]-=a3.z; r[c][15]-=a3.w;
        }
        __syncthreads();
    }

    // quantized = x - residual_final
#pragma unroll
    for (int c = 0; c < 8; ++c) {
        const float* px = x + (size_t)row * RVQ_D + 32 * c + dbase;
        float* pq = qout + (size_t)row * RVQ_D + 32 * c + dbase;
        float4 a0 = *(const float4*)(px);
        float4 a1 = *(const float4*)(px + 4);
        float4 a2 = *(const float4*)(px + 16);
        float4 a3 = *(const float4*)(px + 20);
        float4 o0 = {a0.x - r[c][0],  a0.y - r[c][1],  a0.z - r[c][2],  a0.w - r[c][3]};
        float4 o1 = {a1.x - r[c][4],  a1.y - r[c][5],  a1.z - r[c][6],  a1.w - r[c][7]};
        float4 o2 = {a2.x - r[c][8],  a2.y - r[c][9],  a2.z - r[c][10], a2.w - r[c][11]};
        float4 o3 = {a3.x - r[c][12], a3.y - r[c][13], a3.z - r[c][14], a3.w - r[c][15]};
        *(float4*)(pq)      = o0;
        *(float4*)(pq + 4)  = o1;
        *(float4*)(pq + 16) = o2;
        *(float4*)(pq + 20) = o3;
    }
}

// ---------------------------------------------------------------------------
extern "C" void kernel_launch(void* const* d_in, const int* in_sizes, int n_in,
                              void* d_out, int out_size, void* d_ws, size_t ws_size,
                              hipStream_t stream) {
    const float* x  = (const float*)d_in[0];                    // [N, D]
    const float* cb = (const float*)d_in[1];                    // [L, K, D]

    float* qout = (float*)d_out;                                // [N, D]
    int*   codes = (int*)(qout + (size_t)RVQ_N * RVQ_D);        // [N, L] int32

    // Workspace: bf16 codebooks (4 MB) then c_sq (32 KB).
    __bf16* cbbf = (__bf16*)d_ws;
    float*  csq  = (float*)((char*)d_ws +
                            (size_t)RVQ_LEVELS * RVQ_K * RVQ_D * sizeof(__bf16));

    // 1) Convert codebooks to bf16 + compute squared norms.
    rvq_prep_kernel<<<(RVQ_LEVELS * RVQ_K) / 4, 128, 0, stream>>>(cb, cbbf, csq);

    // 2) Fused residual VQ over all 8 levels.
    rvq_main_kernel<<<RVQ_N / TOK_PER_BLOCK, 256, 0, stream>>>(
        x, cb, cbbf, csq, qout, codes);
}
